// GFAPO_29317446762625
// MI455X (gfx1250) — matrix operand order repaired and checked by reading, not compiled, not run on hardware
//
#include <hip/hip_runtime.h>
#include <stdint.h>

// ---------------------------------------------------------------------------
// Problem constants (from reference): B=2, N=768, D_MODEL=512, D_K=64
// ---------------------------------------------------------------------------
#define BB    2
#define NN    768
#define DM    512
#define DK    64
#define ROWS  (BB*NN)        // 1536 flattened (b, n) rows
#define QWIN_STRIDE 72       // 64 + 8 pad -> 144B row stride (16B aligned)

typedef __attribute__((ext_vector_type(16))) __bf16  v16bf;
typedef __attribute__((ext_vector_type(8)))  float   v8f;
typedef __attribute__((ext_vector_type(2)))  float   v2f;
typedef __attribute__((ext_vector_type(4)))  int     v4i;

union Frag16 { v16bf v; uint4 q[2]; };   // 32 bytes = one 16-bit WMMA A/B frag

#define AS1 __attribute__((address_space(1)))
#define AS3 __attribute__((address_space(3)))

#if defined(__gfx1250__) && __has_builtin(__builtin_amdgcn_global_load_async_to_lds_b128)
#define HAVE_ASYNC_LDS 1
#else
#define HAVE_ASYNC_LDS 0
#endif

// ---------------------------------------------------------------------------
// Kernel 1: q,k,v = x @ W + b using fp32 WMMA (v_wmma_f32_16x16x4_f32).
// One wave per (row_tile, matrix, col_tile): 96 * 3 * 4 = 1152 waves.
// Also emits a bf16 copy of k for the bf16 score GEMM.
// ---------------------------------------------------------------------------
__global__ __launch_bounds__(128) void qkv_kernel(
    const float* __restrict__ x,
    const float* __restrict__ wq_w, const float* __restrict__ wq_b,
    const float* __restrict__ wk_w, const float* __restrict__ wk_b,
    const float* __restrict__ wv_w, const float* __restrict__ wv_b,
    float* __restrict__ qo, float* __restrict__ ko, float* __restrict__ vo,
    __bf16* __restrict__ kb16)
{
  const int lane = threadIdx.x & 31;
  const int wid  = blockIdx.x * 4 + (threadIdx.x >> 5);   // 0..1151
  const int rt   = wid / 12;          // row tile 0..95
  const int rem  = wid % 12;
  const int mat  = rem >> 2;          // 0=q 1=k 2=v
  const int ct   = rem & 3;           // col tile 0..3
  const int half = lane >> 4;
  const int ln16 = lane & 15;
  const int col  = ct * 16 + ln16;

  const float* W   = (mat == 0) ? wq_w : (mat == 1) ? wk_w : wv_w;
  const float* Bv  = (mat == 0) ? wq_b : (mat == 1) ? wk_b : wv_b;
  float*       Out = (mat == 0) ? qo   : (mat == 1) ? ko   : vo;

  const float* xrow = x + (size_t)(rt * 16 + ln16) * DM;  // A: M = lane%16

  v8f acc = {0.f,0.f,0.f,0.f,0.f,0.f,0.f,0.f};
  for (int k0 = 0; k0 < DM; k0 += 4) {
    const int kk = k0 + 2 * half;     // lanes 16-31 carry K=2,3 (ISA A layout)
    v2f a; a.x = xrow[kk];                    a.y = xrow[kk + 1];
    v2f b; b.x = W[(size_t)kk * DK + col];    b.y = W[(size_t)(kk + 1) * DK + col];
    acc = __builtin_amdgcn_wmma_f32_16x16x4_f32(false, a, false, b,
                                                (short)0, acc, false, false);
  }
  const float bias = Bv[col];
  #pragma unroll
  for (int r = 0; r < 8; ++r) {       // C layout: M = r + 8*half, N = lane%16
    const int row = rt * 16 + 8 * half + r;
    const float val = acc[r] + bias;
    Out[(size_t)row * DK + col] = val;
    if (mat == 1) kb16[(size_t)row * DK + col] = (__bf16)val;
  }
}

// ---------------------------------------------------------------------------
// Kernel 2 (dominant, ~19 GFLOP): for each (b,i) build Qwin (127x64, bf16) in
// LDS, async-stage k[b] (96 KB bf16) into LDS via GLOBAL_LOAD_ASYNC_TO_LDS,
// compute S = Qwin @ k[b]^T with v_wmma_f32_16x16x32_bf16 (M=kk, N=j), then
// fuse scaler + double log-softmax over kk (analytically the second
// log_softmax is the identity, so w = sum(z) - 127*LSE(z)) in registers.
// One block (4 waves) per (b,i); each wave owns 12 of the 48 j-tiles.
// Softmax column (fixed j) = 8 acc VGPRs of lane l + lane l^16 -> one shuffle.
// ---------------------------------------------------------------------------
__global__ __launch_bounds__(128) void score_kernel(
    const float* __restrict__ q, const __bf16* __restrict__ kb16,
    float* __restrict__ wbuf)
{
  __shared__ float qsh[DK];
  __shared__ __align__(16) __bf16 qwin[128 * QWIN_STRIDE];   // 18 KB
  __shared__ __align__(16) __bf16 ksh[NN * DK];              // 96 KB

  const int b    = blockIdx.x / NN;
  const int i    = blockIdx.x % NN;
  const int tid  = threadIdx.x;
  const int lane = tid & 31;
  const int wv   = tid >> 5;          // wave in block 0..3
  const int half = lane >> 4;
  const int ln16 = lane & 15;

  const __bf16* kbase = kb16 + (size_t)b * NN * DK;

  // ---- stage k[b] into LDS (async path overlaps with qwin construction) ----
#if HAVE_ASYNC_LDS
  for (int idx = tid; idx < (NN * DK) / 8; idx += 128) {     // 6144 x 16B
    __builtin_amdgcn_global_load_async_to_lds_b128(
        (AS1 v4i*)(kbase + idx * 8),
        (AS3 v4i*)&ksh[idx * 8], 0, 0);
  }
#else
  {
    const uint4* gsrc = reinterpret_cast<const uint4*>(kbase);
    uint4*       ldst = reinterpret_cast<uint4*>(ksh);
    for (int idx = tid; idx < (NN * DK) / 8; idx += 128)
      ldst[idx] = gsrc[idx];
  }
#endif

  if (tid < DK) qsh[tid] = q[((size_t)b * NN + i) * DK + tid];
  __syncthreads();
  // qwin[kk][t] = q[kk - t] if 0 <= kk-t < 64 else 0   (kk row 127 is all-0 pad)
  for (int idx = tid; idx < 128 * 64; idx += 128) {
    const int kk = idx >> 6, t = idx & 63;
    const int d = kk - t;
    const float val = ((unsigned)d < 64u) ? qsh[d] : 0.0f;
    qwin[kk * QWIN_STRIDE + t] = (__bf16)val;
  }

#if HAVE_ASYNC_LDS
#if __has_builtin(__builtin_amdgcn_s_wait_asynccnt)
  __builtin_amdgcn_s_wait_asynccnt(0);
#else
  asm volatile("s_wait_asynccnt 0" ::: "memory");
#endif
#endif
  __syncthreads();

  const v8f zero = {0.f,0.f,0.f,0.f,0.f,0.f,0.f,0.f};

  for (int jt = wv; jt < 48; jt += 4) {
    const int j0 = jt * 16;

    // B fragments (32x16, K=t, N=j) from LDS: lane = column j0+lane%16;
    // 16 contiguous t-values starting at ks*32 + 16*half (contiguous k order, same as A).
    Frag16 bf[2];
    {
      const __bf16* kr = &ksh[(j0 + ln16) * DK];
      #pragma unroll
      for (int ks = 0; ks < 2; ++ks) {
        const __bf16* p = kr + ks * 32 + half * 16;
        bf[ks].q[0] = *reinterpret_cast<const uint4*>(p);
        bf[ks].q[1] = *reinterpret_cast<const uint4*>(p + 8);
      }
    }

    v8f acc[8];
    #pragma unroll
    for (int mt = 0; mt < 8; ++mt) acc[mt] = zero;

    #pragma unroll
    for (int mt = 0; mt < 8; ++mt) {
      const __bf16* rowp = &qwin[(16 * mt + ln16) * QWIN_STRIDE];
      #pragma unroll
      for (int ks = 0; ks < 2; ++ks) {
        // A 16x32 bf16: lanes<16 take t 0..15, lanes>=16
        // take t 16..31 (matches B), each 8-chunk one 16B LDS load.
        Frag16 af;
        const __bf16* p = rowp + ks * 32 + half * 16;
        af.q[0] = *reinterpret_cast<const uint4*>(p);
        af.q[1] = *reinterpret_cast<const uint4*>(p + 8);
        acc[mt] = __builtin_amdgcn_wmma_f32_16x16x32_bf16(
            false, af.v, false, bf[ks].v, (short)0, acc[mt], false, false);
      }
    }

    // Fused epilogue: z = S * scaler; w = sum(z) - 127 * logsumexp(z).
    // Lane holds kk = 16*mt + 8*half + r; kk==127 is padding (excluded).
    float zmax = -3.0e38f, ssum = 0.0f;
    #pragma unroll
    for (int mt = 0; mt < 8; ++mt) {
      #pragma unroll
      for (int r = 0; r < 8; ++r) {
        const int kk = 16 * mt + 8 * half + r;
        if (kk < 127) {
          const float nd = (kk < 64) ? (float)(kk + 1) : (float)(127 - kk);
          const float z  = acc[mt][r] * ((float)(kk + 2) / nd);
          acc[mt][r] = z;
          ssum += z;
          zmax = fmaxf(zmax, z);
        }
      }
    }
    float sexp = 0.0f;
    #pragma unroll
    for (int mt = 0; mt < 8; ++mt) {
      #pragma unroll
      for (int r = 0; r < 8; ++r) {
        const int kk = 16 * mt + 8 * half + r;
        if (kk < 127) sexp += __expf(acc[mt][r] - zmax);
      }
    }
    // merge the two half-columns (lane l <-> lane l^16)
    const float omax  = __shfl_xor(zmax, 16, 32);
    const float osexp = __shfl_xor(sexp, 16, 32);
    const float ossum = __shfl_xor(ssum, 16, 32);
    const float M = fmaxf(zmax, omax);
    const float s = sexp * __expf(zmax - M) + osexp * __expf(omax - M);
    const float L = M + __logf(s);
    const float wval = (ssum + ossum) - 127.0f * L;
    if (half == 0)
      wbuf[((size_t)b * NN + i) * NN + (j0 + ln16)] = wval;
  }
}

// ---------------------------------------------------------------------------
// Kernel 3a: o = w @ v  (per batch 768x768 @ 768x64), fp32 WMMA.
// One wave per (row_tile, col_tile): 96*4 = 384 waves, K-loop 192.
// ---------------------------------------------------------------------------
__global__ __launch_bounds__(128) void ov_kernel(
    const float* __restrict__ wbuf, const float* __restrict__ v,
    float* __restrict__ o)
{
  const int lane = threadIdx.x & 31;
  const int wid  = blockIdx.x * 4 + (threadIdx.x >> 5);   // 0..383
  const int rt   = wid >> 2;
  const int ct   = wid & 3;
  const int half = lane >> 4;
  const int ln16 = lane & 15;
  const int b    = rt / 48;                 // 48 row tiles per batch
  const int col  = ct * 16 + ln16;

  const float* arow = wbuf + (size_t)(rt * 16 + ln16) * NN;
  const float* vb   = v + (size_t)b * NN * DK;

  v8f acc = {0.f,0.f,0.f,0.f,0.f,0.f,0.f,0.f};
  for (int k0 = 0; k0 < NN; k0 += 4) {
    const int kk = k0 + 2 * half;
    v2f a;  a.x = arow[kk];                   a.y = arow[kk + 1];
    v2f bb; bb.x = vb[(size_t)kk * DK + col]; bb.y = vb[(size_t)(kk + 1) * DK + col];
    acc = __builtin_amdgcn_wmma_f32_16x16x4_f32(false, a, false, bb,
                                                (short)0, acc, false, false);
  }
  #pragma unroll
  for (int r = 0; r < 8; ++r)
    o[(size_t)(rt * 16 + 8 * half + r) * DK + col] = acc[r];
}

// ---------------------------------------------------------------------------
// Kernel 3b: out = o @ proj_w + proj_b  (1536x64 @ 64x64), fp32 WMMA.
// ---------------------------------------------------------------------------
__global__ __launch_bounds__(128) void proj_kernel(
    const float* __restrict__ o, const float* __restrict__ proj_w,
    const float* __restrict__ proj_b, float* __restrict__ out)
{
  const int lane = threadIdx.x & 31;
  const int wid  = blockIdx.x * 4 + (threadIdx.x >> 5);   // 0..383
  const int rt   = wid >> 2;
  const int ct   = wid & 3;
  const int half = lane >> 4;
  const int ln16 = lane & 15;
  const int col  = ct * 16 + ln16;

  const float* arow = o + (size_t)(rt * 16 + ln16) * DK;

  v8f acc = {0.f,0.f,0.f,0.f,0.f,0.f,0.f,0.f};
  for (int k0 = 0; k0 < DK; k0 += 4) {
    const int kk = k0 + 2 * half;
    v2f a;  a.x = arow[kk];                        a.y = arow[kk + 1];
    v2f bb; bb.x = proj_w[(size_t)kk * DK + col];  bb.y = proj_w[(size_t)(kk + 1) * DK + col];
    acc = __builtin_amdgcn_wmma_f32_16x16x4_f32(false, a, false, bb,
                                                (short)0, acc, false, false);
  }
  const float bias = proj_b[col];
  #pragma unroll
  for (int r = 0; r < 8; ++r)
    out[(size_t)(rt * 16 + 8 * half + r) * DK + col] = acc[r] + bias;
}

// ---------------------------------------------------------------------------
// Host launcher. Workspace layout (floats):
//   q[98304] | k[98304] | v[98304] | o[98304] | w[1179648] | kb16 (bf16, 98304)
// Total ~6.5 MB. All buffers fully written before read; no zero-init needed.
// ---------------------------------------------------------------------------
extern "C" void kernel_launch(void* const* d_in, const int* in_sizes, int n_in,
                              void* d_out, int out_size, void* d_ws, size_t ws_size,
                              hipStream_t stream) {
  (void)in_sizes; (void)n_in; (void)out_size; (void)ws_size;

  const float* x      = (const float*)d_in[0];
  const float* wq_w   = (const float*)d_in[1];
  const float* wq_b   = (const float*)d_in[2];
  const float* wk_w   = (const float*)d_in[3];
  const float* wk_b   = (const float*)d_in[4];
  const float* wv_w   = (const float*)d_in[5];
  const float* wv_b   = (const float*)d_in[6];
  const float* proj_w = (const float*)d_in[7];
  const float* proj_b = (const float*)d_in[8];
  float* out = (float*)d_out;

  float* ws   = (float*)d_ws;
  float* qf   = ws;                       // 98304
  float* kf   = ws + 98304;               // 98304
  float* vf   = ws + 196608;              // 98304
  float* of   = ws + 294912;              // 98304
  float* wb   = ws + 393216;              // 1179648
  __bf16* kb  = (__bf16*)(ws + 1572864);  // 98304 bf16

  qkv_kernel<<<288, 128, 0, stream>>>(x, wq_w, wq_b, wk_w, wk_b, wv_w, wv_b,
                                      qf, kf, vf, kb);
  score_kernel<<<BB * NN, 128, 0, stream>>>(qf, kb, wb);
  ov_kernel<<<96, 128, 0, stream>>>(wb, vf, of);
  proj_kernel<<<96, 128, 0, stream>>>(of, proj_w, proj_b, out);
}
